// RelationMessagePassingModel_56788057588329
// MI455X (gfx1250) — matrix operand order repaired and checked by reading, not compiled, not run on hardware
//
#include <hip/hip_runtime.h>
#include <hip/hip_bf16.h>

// ---------------- problem constants (from reference) ----------------
#define E      64
#define NOBJ   131072
#define NGRAPH 64
#define NPG    2048
#define SHARPF 12.0f
#define INIT_KEY 0x007FFFFFu   // ordered-uint key of -inf

typedef __attribute__((ext_vector_type(16))) __bf16 v16bf;
typedef __attribute__((ext_vector_type(8)))  __bf16 v8bf;
typedef __attribute__((ext_vector_type(8)))  float  v8f;

__device__ inline v8f wmma_bf16(v16bf a, v16bf b, v8f c) {
  // (neg_a, A, neg_b, B, c_mod, C, reuse_a, reuse_b)
  return __builtin_amdgcn_wmma_f32_16x16x32_bf16(false, a, false, b, (short)0, c,
                                                 false, false);
}

// Fast mish: with t=e^x, tanh(softplus(x)) = 1 - 2/(t^2+2t+2)
//   mish(x) = x - 2x * rcp(fma(t, t+2, 2))
// Saturates correctly: t^2 -> inf => rcp -> 0 => mish -> x; t -> 0 => mish -> 0.
// One v_exp_f32 + one v_rcp_f32 + ~4 FMA/MUL (vs ~80-op libm tanh/log1p chain).
__device__ inline float mishf(float x) {
  float t = __expf(x);
  float d = __builtin_fmaf(t, t + 2.0f, 2.0f);
  return x - 2.0f * x * __builtin_amdgcn_rcpf(d);
}

// order-preserving float <-> uint for atomic max
__device__ inline unsigned fkey(float f) {
  unsigned u = __float_as_uint(f);
  return (u & 0x80000000u) ? ~u : (u | 0x80000000u);
}
__device__ inline float kval(unsigned k) {
  unsigned u = (k & 0x80000000u) ? (k & 0x7FFFFFFFu) : ~k;
  return __uint_as_float(u);
}

// A fragment, 16-bit A-matrix 16x32 layout (ISA 7.12.2):
// lane l -> row M=l%16; group=l/16; elements 0..7 = K(group*8 + e),
// elements 8..15 = K(16 + group*8 + e). Two 16B LDS loads per fragment.
__device__ inline v16bf load_a(const __bf16* p, int group) {
  v8bf lo = *(const v8bf*)(p + group * 8);
  v8bf hi = *(const v8bf*)(p + 16 + group * 8);
  v16bf a;
#pragma unroll
  for (int i = 0; i < 8; ++i) { a[i] = lo[i]; a[i + 8] = hi[i]; }
  return a;
}

// -------------------------------------------------------------------
// Generic fused MLP kernel:  y = (x + mish(x@Wr + br)) @ Wo + bo
// 256 threads = 8 waves, 128 rows per block, wave owns 16 rows.
// MODE 0: relation pass1 (atomic segment max of messages)
// MODE 1: relation pass2 (exp + atomic segment sum)
// MODE 2: node update   (x = [max_msg, emb]) -> out
// MODE 3: readout value (x = agg)            -> out
// MODE 4: readout update(x = [emb, r[g]])    -> out
// -------------------------------------------------------------------
template <int MODE, int ARITY, int DIN, int DOUT>
__global__ __launch_bounds__(256) void mlp_kernel(
    const float* __restrict__ src, const float* __restrict__ aux,
    const int* __restrict__ idx,
    const __bf16* __restrict__ WrS, const float* __restrict__ br,
    const __bf16* __restrict__ WoS, const float* __restrict__ bo,
    unsigned* __restrict__ mkey, float* __restrict__ ssum,
    float* __restrict__ out, int nrows) {
  __shared__ __align__(16) __bf16 xlds[128 * DIN];
  __shared__ __align__(16) __bf16 ulds[128 * DIN];
  const int tid  = threadIdx.x;
  const int row0 = blockIdx.x * 128;
  const float* mflt = (const float*)mkey;  // after finalize, holds float m

  // ---- cooperative gather: build x rows (bf16) in LDS ----
  for (int e = tid; e < 128 * DIN; e += 256) {
    int rl  = e / DIN;
    int col = e - rl * DIN;
    int row = row0 + rl;
    int rr  = row < nrows ? row : nrows - 1;  // clamp (epilogue guards writes)
    float v;
    if constexpr (MODE == 0 || MODE == 1) {
      int j = (ARITY == 2) ? (col >> 6) : 0;
      v = src[(size_t)idx[rr * ARITY + j] * E + (col & 63)];
    } else if constexpr (MODE == 2) {
      v = (col < E)
              ? (__logf(ssum[(size_t)rr * E + col] + 1e-16f) * (1.0f / SHARPF) +
                 mflt[(size_t)rr * E + col])
              : src[(size_t)rr * E + (col - E)];
    } else if constexpr (MODE == 3) {
      v = src[(size_t)rr * DIN + col];
    } else {  // MODE 4
      v = (col < E) ? src[(size_t)rr * E + col]
                    : aux[(size_t)(rr >> 11) * E + (col - E)];
    }
    xlds[e] = (__bf16)v;
  }
  __syncthreads();

  const int wave = tid >> 5, lane = tid & 31;
  const int group = lane >> 4, lrow = lane & 15;
  const __bf16* xw = xlds + wave * 16 * DIN;
  __bf16* uw = ulds + wave * 16 * DIN;
  constexpr int KT = DIN / 32;

  // ---- hidden GEMM + mish + residual -> u (bf16 in LDS) ----
#pragma unroll
  for (int nt = 0; nt < DIN / 16; ++nt) {
    v8f acc = {};
#pragma unroll
    for (int kt = 0; kt < KT; ++kt) {
      v16bf a = load_a(xw + lrow * DIN + kt * 32, group);
      v16bf b = *(const v16bf*)(WrS + (size_t)((nt * KT + kt) * 32 + lane) * 16);
      acc = wmma_bf16(a, b, acc);
    }
    int col = nt * 16 + lrow;
    float bb = br[col];
#pragma unroll
    for (int v = 0; v < 8; ++v) {
      int r = v + group * 8;  // C/D layout: vgpr v, lanes 16-31 -> M+8
      float u = (float)xw[r * DIN + col] + mishf(acc[v] + bb);
      uw[r * DIN + col] = (__bf16)u;
    }
  }
  __syncthreads();

  // ---- output GEMM + per-mode epilogue ----
#pragma unroll
  for (int nt = 0; nt < DOUT / 16; ++nt) {
    v8f acc = {};
#pragma unroll
    for (int kt = 0; kt < KT; ++kt) {
      v16bf a = load_a(uw + lrow * DIN + kt * 32, group);
      v16bf b = *(const v16bf*)(WoS + (size_t)((nt * KT + kt) * 32 + lane) * 16);
      acc = wmma_bf16(a, b, acc);
    }
    int col = nt * 16 + lrow;
    float bb = bo[col];
#pragma unroll
    for (int v = 0; v < 8; ++v) {
      int rl  = wave * 16 + v + group * 8;
      int row = row0 + rl;
      if (row >= nrows) continue;
      float y = acc[v] + bb;
      if constexpr (MODE == 0) {
        int tgt = idx[row * ARITY + ((ARITY == 2) ? (col >> 6) : 0)];
        atomicMax(&mkey[(size_t)tgt * E + (col & 63)], fkey(y));
      } else if constexpr (MODE == 1) {
        int tgt = idx[row * ARITY + ((ARITY == 2) ? (col >> 6) : 0)];
        size_t o = (size_t)tgt * E + (col & 63);
        atomicAdd(&ssum[o], __expf(SHARPF * (y - mflt[o])));
      } else {
        out[(size_t)row * DOUT + col] = y;
      }
    }
  }
}

// Pre-swizzle a row-major fp32 weight (K x doutw) into bf16 B-fragment order:
// dst[((nt*KT+kt)*32 + lane)*16 + e] = W[(kt*32 + (e/8)*16 + (lane/16)*8 + e%8)*doutw + nt*16 + lane%16]
__global__ void swizzle_w_kernel(const float* __restrict__ W,
                                 __bf16* __restrict__ dst, int doutw, int KT,
                                 int total) {
  int t = blockIdx.x * 256 + threadIdx.x;
  if (t >= total) return;
  int e    = t & 15;
  int lane = (t >> 4) & 31;
  int kt   = (t >> 9) % KT;
  int nt   = t / (512 * KT);
  int k = kt * 32 + (e >> 3) * 16 + (lane >> 4) * 8 + (e & 7);
  int n = nt * 16 + (lane & 15);
  dst[t] = (__bf16)W[(size_t)k * doutw + n];
}

__global__ void init_agg_kernel(unsigned* __restrict__ mkey,
                                float* __restrict__ ssum, int n) {
  int i = blockIdx.x * 256 + threadIdx.x;
  if (i < n) { mkey[i] = INIT_KEY; ssum[i] = 0.0f; }
}

// decode key -> float m, with untouched rows (-inf) -> 0, in place
__global__ void finalize_m_kernel(unsigned* __restrict__ mkey, int n) {
  int i = blockIdx.x * 256 + threadIdx.x;
  if (i < n) {
    unsigned k = mkey[i];
    float m = (k == INIT_KEY) ? 0.0f : kval(k);
    ((float*)mkey)[i] = m;
  }
}

// per-graph sum over 2048 contiguous objects
__global__ void graph_sum_kernel(const float* __restrict__ emb,
                                 float* __restrict__ agg) {
  __shared__ float red[256];
  int g = blockIdx.x, tid = threadIdx.x;
  int col = tid & 63, chunk = tid >> 6;  // 4 chunks of 512 rows
  const float* base = emb + (size_t)g * NPG * E;
  float s = 0.0f;
  for (int o = chunk * (NPG / 4); o < (chunk + 1) * (NPG / 4); ++o)
    s += base[(size_t)o * E + col];
  red[tid] = s;
  __syncthreads();
  if (tid < 128) red[tid] += red[tid + 128];
  __syncthreads();
  if (tid < 64) agg[(size_t)g * E + tid] = red[tid] + red[tid + 64];
}

// -------------------------------------------------------------------
extern "C" void kernel_launch(void* const* d_in, const int* in_sizes, int n_in,
                              void* d_out, int out_size, void* d_ws,
                              size_t ws_size, hipStream_t stream) {
  const float* emb0 = (const float*)d_in[0];
  const int* rel0 = (const int*)d_in[1];
  const int* rel1 = (const int*)d_in[2];
  const int* rel2 = (const int*)d_in[3];
  const int n0 = in_sizes[1];      // 300000 atoms (arity 1)
  const int n1 = in_sizes[2] / 2;  // 300000 tuples (arity 2)
  const int n2 = in_sizes[3] / 2;

  // ---- workspace carve (~135 MB) ----
  char* ws = (char*)d_ws;
  __bf16* Wb = (__bf16*)ws;             // swizzled bf16 weights (256 KB)
  size_t off = (size_t)1 << 20;
  unsigned* mkey = (unsigned*)(ws + off); off += (size_t)NOBJ * E * 4;  // 32 MB
  float* ssum = (float*)(ws + off);       off += (size_t)NOBJ * E * 4;  // 32 MB
  float* B1   = (float*)(ws + off);       off += (size_t)NOBJ * E * 4;  // 32 MB
  float* B2   = (float*)(ws + off);       off += (size_t)NOBJ * E * 4;  // 32 MB
  float* agg  = (float*)(ws + off);       off += (size_t)NGRAPH * E * 4;
  float* rbuf = (float*)(ws + off);

  // ---- convert+swizzle all weights to bf16 fragment order ----
  struct MatDesc { int widx; int off; int din; int dout; };
  static const MatDesc mats[12] = {
      {4, 0, 64, 64},        {6, 4096, 64, 64},       // Wr0, Wo0
      {8, 8192, 128, 128},   {10, 24576, 128, 128},   // Wr1, Wo1
      {12, 40960, 128, 128}, {14, 57344, 128, 128},   // Wr2, Wo2
      {16, 73728, 128, 128}, {18, 90112, 128, 64},    // Wru, Wou
      {20, 98304, 64, 64},   {22, 102400, 64, 64},    // Wrv, Wov
      {24, 106496, 128, 128},{26, 122880, 128, 64},   // Wrr, Wor
  };
  for (int i = 0; i < 12; ++i) {
    int total = mats[i].din * mats[i].dout;
    swizzle_w_kernel<<<(total + 255) / 256, 256, 0, stream>>>(
        (const float*)d_in[mats[i].widx], Wb + mats[i].off, mats[i].dout,
        mats[i].din / 32, total);
  }
  const __bf16 *Wr0 = Wb + 0,     *Wo0 = Wb + 4096,  *Wr1 = Wb + 8192,
               *Wo1 = Wb + 24576, *Wr2 = Wb + 40960, *Wo2 = Wb + 57344,
               *Wru = Wb + 73728, *Wou = Wb + 90112, *Wrv = Wb + 98304,
               *Wov = Wb + 102400,*Wrr = Wb + 106496,*Wor = Wb + 122880;
  const float *br0 = (const float*)d_in[5],  *bo0 = (const float*)d_in[7],
              *br1 = (const float*)d_in[9],  *bo1 = (const float*)d_in[11],
              *br2 = (const float*)d_in[13], *bo2 = (const float*)d_in[15],
              *bru = (const float*)d_in[17], *bou = (const float*)d_in[19],
              *brv = (const float*)d_in[21], *bov = (const float*)d_in[23],
              *brr = (const float*)d_in[25], *bor = (const float*)d_in[27];

  const int nagg = NOBJ * E;
  const float* cur = emb0;
  for (int layer = 0; layer < 2; ++layer) {
    init_agg_kernel<<<(nagg + 255) / 256, 256, 0, stream>>>(mkey, ssum, nagg);
    // pass 1: segment max (messages recomputed, not materialized)
    mlp_kernel<0, 1, 64, 64><<<(n0 + 127) / 128, 256, 0, stream>>>(
        cur, nullptr, rel0, Wr0, br0, Wo0, bo0, mkey, ssum, nullptr, n0);
    mlp_kernel<0, 2, 128, 128><<<(n1 + 127) / 128, 256, 0, stream>>>(
        cur, nullptr, rel1, Wr1, br1, Wo1, bo1, mkey, ssum, nullptr, n1);
    mlp_kernel<0, 2, 128, 128><<<(n2 + 127) / 128, 256, 0, stream>>>(
        cur, nullptr, rel2, Wr2, br2, Wo2, bo2, mkey, ssum, nullptr, n2);
    finalize_m_kernel<<<(nagg + 255) / 256, 256, 0, stream>>>(mkey, nagg);
    // pass 2: exp + segment sum
    mlp_kernel<1, 1, 64, 64><<<(n0 + 127) / 128, 256, 0, stream>>>(
        cur, nullptr, rel0, Wr0, br0, Wo0, bo0, mkey, ssum, nullptr, n0);
    mlp_kernel<1, 2, 128, 128><<<(n1 + 127) / 128, 256, 0, stream>>>(
        cur, nullptr, rel1, Wr1, br1, Wo1, bo1, mkey, ssum, nullptr, n1);
    mlp_kernel<1, 2, 128, 128><<<(n2 + 127) / 128, 256, 0, stream>>>(
        cur, nullptr, rel2, Wr2, br2, Wo2, bo2, mkey, ssum, nullptr, n2);
    // node update: emb_mid = mlp([max_msg, emb])
    mlp_kernel<2, 1, 128, 64><<<NOBJ / 128, 256, 0, stream>>>(
        cur, nullptr, nullptr, Wru, bru, Wou, bou, mkey, ssum, B1, NOBJ);
    // per-graph readout
    graph_sum_kernel<<<NGRAPH, 256, 0, stream>>>(B1, agg);
    mlp_kernel<3, 1, 64, 64><<<1, 256, 0, stream>>>(
        agg, nullptr, nullptr, Wrv, brv, Wov, bov, mkey, ssum, rbuf, NGRAPH);
    // readout update: emb_next = mlp([emb_mid, r[graph]])
    float* nxt = (layer == 0) ? B2 : (float*)d_out;
    mlp_kernel<4, 1, 128, 64><<<NOBJ / 128, 256, 0, stream>>>(
        B1, rbuf, nullptr, Wrr, brr, Wor, bor, mkey, ssum, nxt, NOBJ);
    cur = nxt;
  }
}
